// LEquiTFRtModel_73744588472517
// MI455X (gfx1250) — matrix operand order repaired and checked by reading, not compiled, not run on hardware
//
#include <hip/hip_runtime.h>
#include <hip/hip_bf16.h>
#include <math.h>

// ---------------------------------------------------------------------------
// CDNA5 (gfx1250) types & helpers
// ---------------------------------------------------------------------------
typedef __attribute__((ext_vector_type(16))) __bf16 v16bf;
typedef __attribute__((ext_vector_type(8)))  float  v8f;

union FragBF { v16bf v; unsigned int u[8]; };

// native f32 -> bf16 conversion (lets clang emit hardware cvt ops)
__device__ __forceinline__ unsigned short f2bf(float f) {
    union { __bf16 h; unsigned short s; } x;
    x.h = (__bf16)f;
    return x.s;
}
__device__ __forceinline__ unsigned int pack2(float a, float b) {
    union { __bf16 h[2]; unsigned int u; } x;
    x.h[0] = (__bf16)a; x.h[1] = (__bf16)b;
    return x.u;
}
__device__ __forceinline__ float bf2f(unsigned short u) {
    return __uint_as_float(((unsigned int)u) << 16);
}

// ---------------------------------------------------------------------------
// GEMM: C[M,N] = act(A[M,K] x W + bias) (+ residual)
//   A:  f32 [M,K]  (converted to bf16 while staging into LDS)
//   WT: bf16, PRE-TRANSPOSED to [N,K]; B tiles staged with
//       global_load_async_to_lds_b128 (ASYNCcnt), double-buffered so the
//       async engine and the A-stage VALU overlap the 4 WMMAs of the
//       previous tile. One barrier per K-step.
//   Requires K%64==0 and N%64==0 (true for every matmul in this model).
// act: 0=none, 1=GELU(exact), 2=SiLU.  obf: 1 -> store bf16 output.
// ---------------------------------------------------------------------------
__global__ __launch_bounds__(256) void gemm_kernel(
    const float* __restrict__ A, const unsigned short* __restrict__ WT,
    const float* __restrict__ bias, const float* __restrict__ residual,
    void* __restrict__ Cout, int M, int N, int K, int act, int obf)
{
    __shared__ __align__(16) unsigned short As[2][64 * 72];
    __shared__ __align__(16) unsigned short Bs[2][64 * 72];

    int tid  = threadIdx.x;
    int lane = tid & 31, wave = tid >> 5;
    int wm = wave & 3, wn = wave >> 2;
    int m0 = blockIdx.y * 64, n0 = blockIdx.x * 64;
    int half = lane >> 4, l15 = lane & 15;

    // staging coordinates: each thread owns a 16-element column group of one row
    int ar = tid >> 2;            // 0..63 : A row / B row(n)
    int ac = (tid & 3) * 16;
    const bool mfull = (m0 + 64 <= M);
    const bool arok  = (m0 + ar) < M;
    const float* Arow = A + (size_t)(m0 + ar) * K + ac;
    const unsigned short* Brow = WT + (size_t)(n0 + ar) * K + ac;
    unsigned ldsB0 = (unsigned)(size_t)&Bs[0][ar * 72 + ac];
    const unsigned bufBytes = 64u * 72u * 2u;

    auto issueB = [&](int k0, int buf) {
        const unsigned short* g = Brow + k0;
        unsigned l = ldsB0 + (unsigned)buf * bufBytes;
        asm volatile("global_load_async_to_lds_b128 %0, %1, off"
                     :: "v"(l), "v"(g) : "memory");
        asm volatile("global_load_async_to_lds_b128 %0, %1, off"
                     :: "v"(l + 16u), "v"(g + 8) : "memory");
    };
    auto stageA = [&](int k0, int buf) {
        const float4* p = (const float4*)(Arow + k0);
        unsigned short* dst = &As[buf][ar * 72 + ac];
        if (mfull) {
            #pragma unroll
            for (int j = 0; j < 4; j++) {
                float4 f4 = p[j];
                uint2 pr; pr.x = pack2(f4.x, f4.y); pr.y = pack2(f4.z, f4.w);
                *(uint2*)(dst + 4 * j) = pr;
            }
        } else {
            #pragma unroll
            for (int j = 0; j < 4; j++) {
                float4 f4 = arok ? p[j] : (float4){0.f, 0.f, 0.f, 0.f};
                uint2 pr; pr.x = pack2(f4.x, f4.y); pr.y = pack2(f4.z, f4.w);
                *(uint2*)(dst + 4 * j) = pr;
            }
        }
    };

    v8f acc0 = {}, acc1 = {};
    int arow = wm * 16 + l15;
    int bcol0 = wn * 32 + l15, bcol1 = bcol0 + 16;
    int nk = K >> 6;

    // prologue: tile 0 into buffer 0
    issueB(0, 0);
    stageA(0, 0);
    asm volatile("s_wait_asynccnt 0x0" ::: "memory");
    __syncthreads();

    for (int i = 0; i < nk; i++) {
        int cb = i & 1, nb = cb ^ 1;
        if (i + 1 < nk) {               // prefetch next tile into other buffer
            issueB((i + 1) << 6, nb);
            stageA((i + 1) << 6, nb);
        }
        // compute current tile: 2 K-substeps x 2 N-subtiles = 4 WMMA / wave
        #pragma unroll
        for (int ks = 0; ks < 2; ks++) {
            FragBF a, b0, b1;
            int koff = ks * 32;
            #pragma unroll
            for (int r = 0; r < 8; r++) {
                int ka = koff + 2 * r + (half << 4);
                a.u[r] = *(const unsigned int*)&As[cb][arow * 72 + ka];
                int kb = koff + 2 * r + (half << 4);
                b0.u[r] = *(const unsigned int*)&Bs[cb][bcol0 * 72 + kb];
                b1.u[r] = *(const unsigned int*)&Bs[cb][bcol1 * 72 + kb];
            }
            acc0 = __builtin_amdgcn_wmma_f32_16x16x32_bf16(false, a.v, false, b0.v,
                                                           (short)0, acc0, false, false);
            acc1 = __builtin_amdgcn_wmma_f32_16x16x32_bf16(false, a.v, false, b1.v,
                                                           (short)0, acc1, false, false);
        }
        asm volatile("s_wait_asynccnt 0x0" ::: "memory");
        __syncthreads();
    }

    // ---- epilogue ----
    int crow0 = m0 + wm * 16 + (half << 3);
    int ccol0 = n0 + wn * 32 + l15;
    int ccol1 = ccol0 + 16;
    float bv0 = bias ? bias[ccol0] : 0.f;
    float bv1 = bias ? bias[ccol1] : 0.f;
    #pragma unroll
    for (int r = 0; r < 8; r++) {
        int row = crow0 + r;
        if (row < M) {
            float v0 = acc0[r] + bv0;
            float v1 = acc1[r] + bv1;
            if (act == 1) {
                v0 = 0.5f * v0 * (1.f + erff(v0 * 0.70710678118654752f));
                v1 = 0.5f * v1 * (1.f + erff(v1 * 0.70710678118654752f));
            } else if (act == 2) {
                v0 = v0 / (1.f + __expf(-v0));
                v1 = v1 / (1.f + __expf(-v1));
            }
            if (obf) {
                unsigned short* Cb = (unsigned short*)Cout;
                Cb[(size_t)row * N + ccol0] = f2bf(v0);
                Cb[(size_t)row * N + ccol1] = f2bf(v1);
            } else {
                float* Cf = (float*)Cout;
                if (residual) {
                    v0 += residual[(size_t)row * N + ccol0];
                    v1 += residual[(size_t)row * N + ccol1];
                }
                Cf[(size_t)row * N + ccol0] = v0;
                Cf[(size_t)row * N + ccol1] = v1;
            }
        }
    }
}

// ---------------------------------------------------------------------------
// Flash attention (bf16 Q/K/V, f32 out). One wave per (16-query tile, head, b).
// Online softmax in registers; P tile round-trips LDS (bf16) to reach WMMA
// A-fragment layout. mode 0: causal; mode 1: joint readout mask.
// ---------------------------------------------------------------------------
__device__ __forceinline__ bool attn_allowed(int q, int k, int mode, int D) {
    if (mode == 0) return k <= q;
    return (k < D) ? (k <= q) : ((k - D) >= q);
}

__global__ __launch_bounds__(32) void attn_kernel(
    const unsigned short* __restrict__ Qp, const unsigned short* __restrict__ Kp,
    const unsigned short* __restrict__ Vp, float* __restrict__ Op,
    int TQ, int TK,
    int qrs, long long qbs, int krs, long long kbs,
    int vrs, long long vbs, int ors, long long obs,
    float scale, int mode, int maskD)
{
    __shared__ __align__(8) unsigned short Pt[16 * 32];

    int lane = threadIdx.x;
    int h = blockIdx.y, b = blockIdx.z;
    int qbase = blockIdx.x * 16;
    int half = lane >> 4, l15 = lane & 15;

    const unsigned short* Q = Qp + (long long)b * qbs + h * 64;
    const unsigned short* K = Kp + (long long)b * kbs + h * 64;
    const unsigned short* V = Vp + (long long)b * vbs + h * 64;
    float*                O = Op + (long long)b * obs + h * 64;

    FragBF a0, a1;
    {
        int qm = qbase + l15;
        bool ok = qm < TQ;
        const unsigned short* qr = Q + (long long)qm * qrs;
        #pragma unroll
        for (int r = 0; r < 8; r++) {
            int kd = 2 * r + (half << 4);
            a0.u[r] = ok ? *(const unsigned int*)&qr[kd]      : 0u;
            a1.u[r] = ok ? *(const unsigned int*)&qr[32 + kd] : 0u;
        }
    }

    v8f o0 = {}, o1 = {}, o2 = {}, o3 = {};
    float mrow[8], lrow[8];
    #pragma unroll
    for (int r = 0; r < 8; r++) { mrow[r] = -1e30f; lrow[r] = 0.f; }

    int qrow0 = qbase + (half << 3);
    int ntiles = (TK + 31) / 32;

    for (int kt = 0; kt < ntiles; kt++) {
        int kb = kt * 32;
        v8f s0 = {}, s1 = {};

        #pragma unroll
        for (int ch = 0; ch < 2; ch++) {
            FragBF bk0, bk1;
            int key0 = kb + l15, key1 = kb + 16 + l15;
            bool ok0 = key0 < TK, ok1 = key1 < TK;
            const unsigned short* kr0 = K + (long long)key0 * krs + ch * 32;
            const unsigned short* kr1 = K + (long long)key1 * krs + ch * 32;
            #pragma unroll
            for (int r = 0; r < 8; r++) {
                int kd = 2 * r + (half << 4);
                bk0.u[r] = ok0 ? *(const unsigned int*)&kr0[kd] : 0u;
                bk1.u[r] = ok1 ? *(const unsigned int*)&kr1[kd] : 0u;
            }
            const FragBF& aa = ch ? a1 : a0;
            s0 = __builtin_amdgcn_wmma_f32_16x16x32_bf16(false, aa.v, false, bk0.v,
                                                         (short)0, s0, false, false);
            s1 = __builtin_amdgcn_wmma_f32_16x16x32_bf16(false, aa.v, false, bk1.v,
                                                         (short)0, s1, false, false);
        }

        int kcol0 = kb + l15, kcol1 = kb + 16 + l15;
        #pragma unroll
        for (int r = 0; r < 8; r++) {
            int q = qrow0 + r;
            float v0 = s0[r] * scale, v1 = s1[r] * scale;
            if (kcol0 >= TK || !attn_allowed(q, kcol0, mode, maskD)) v0 = -1e30f;
            if (kcol1 >= TK || !attn_allowed(q, kcol1, mode, maskD)) v1 = -1e30f;
            float mx = fmaxf(v0, v1);
            #pragma unroll
            for (int d = 1; d < 16; d <<= 1) mx = fmaxf(mx, __shfl_xor(mx, d));
            float mnew = fmaxf(mrow[r], mx);
            float corr = __expf(mrow[r] - mnew);
            float p0 = __expf(v0 - mnew), p1 = __expf(v1 - mnew);
            float rs = p0 + p1;
            #pragma unroll
            for (int d = 1; d < 16; d <<= 1) rs += __shfl_xor(rs, d);
            lrow[r] = lrow[r] * corr + rs;
            mrow[r] = mnew;
            o0[r] *= corr; o1[r] *= corr; o2[r] *= corr; o3[r] *= corr;
            int prow = r + (half << 3);
            Pt[prow * 32 + l15]      = f2bf(p0);
            Pt[prow * 32 + 16 + l15] = f2bf(p1);
        }
        __syncthreads();

        FragBF pa;
        #pragma unroll
        for (int r = 0; r < 8; r++) {
            int kd = 2 * r + (half << 4);
            pa.u[r] = *(const unsigned int*)&Pt[l15 * 32 + kd];
        }
        #pragma unroll
        for (int nt = 0; nt < 4; nt++) {
            FragBF bv;
            int hd = nt * 16 + l15;
            #pragma unroll
            for (int r = 0; r < 8; r++) {
                int key = kb + 2 * r + (half << 4);
                unsigned int u0 = (key     < TK) ? (unsigned int)V[(long long)key * vrs + hd]       : 0u;
                unsigned int u1 = (key + 1 < TK) ? (unsigned int)V[(long long)(key + 1) * vrs + hd] : 0u;
                bv.u[r] = u0 | (u1 << 16);
            }
            v8f& oo = nt == 0 ? o0 : nt == 1 ? o1 : nt == 2 ? o2 : o3;
            oo = __builtin_amdgcn_wmma_f32_16x16x32_bf16(false, pa.v, false, bv.v,
                                                         (short)0, oo, false, false);
        }
        __syncthreads();
    }

    #pragma unroll
    for (int r = 0; r < 8; r++) {
        int q = qrow0 + r;
        if (q < TQ && lrow[r] > 0.f) {
            float inv = 1.f / lrow[r];
            O[(long long)q * ors +      l15] = o0[r] * inv;
            O[(long long)q * ors + 16 + l15] = o1[r] * inv;
            O[(long long)q * ors + 32 + l15] = o2[r] * inv;
            O[(long long)q * ors + 48 + l15] = o3[r] * inv;
        }
    }
}

// ---------------------------------------------------------------------------
// LayerNorm: one block (256 threads) per row of length C.
// ---------------------------------------------------------------------------
__global__ __launch_bounds__(256) void layernorm_kernel(
    const float* __restrict__ x, const float* __restrict__ g,
    const float* __restrict__ bta, float* __restrict__ y, int C)
{
    __shared__ float red[256];
    int row = blockIdx.x, tid = threadIdx.x;
    const float* xr = x + (size_t)row * C;
    float local[4]; int cnt = 0; float s = 0.f;
    for (int i = tid; i < C; i += 256) { float v = xr[i]; local[cnt++] = v; s += v; }
    red[tid] = s; __syncthreads();
    for (int st = 128; st > 0; st >>= 1) { if (tid < st) red[tid] += red[tid + st]; __syncthreads(); }
    float mean = red[0] / (float)C; __syncthreads();
    float s2 = 0.f;
    for (int j = 0; j < cnt; j++) { float d = local[j] - mean; s2 += d * d; }
    red[tid] = s2; __syncthreads();
    for (int st = 128; st > 0; st >>= 1) { if (tid < st) red[tid] += red[tid + st]; __syncthreads(); }
    float inv = rsqrtf(red[0] / (float)C + 1e-5f);
    cnt = 0;
    for (int i = tid; i < C; i += 256)
        y[(size_t)row * C + i] = (local[cnt++] - mean) * inv * g[i] + bta[i];
}

// ---------------------------------------------------------------------------
// Weight pre-conversion: f32 [K,N] -> bf16 [N,K] (transpose), or direct copy.
// ---------------------------------------------------------------------------
__global__ void wconv_kernel(const float* __restrict__ w, unsigned short* __restrict__ wt,
                             int K, int N)
{
    int i = blockIdx.x * blockDim.x + threadIdx.x;
    if (i >= K * N) return;
    int n = i % N, k = i / N;                  // coalesced read
    wt[(size_t)n * K + k] = f2bf(w[i]);
}
__global__ void conv_kernel(const float* __restrict__ s, unsigned short* __restrict__ d, int n)
{
    int i = blockIdx.x * blockDim.x + threadIdx.x;
    if (i < n) d[i] = f2bf(s[i]);
}

// ---------------------------------------------------------------------------
// Elementwise kernels
// ---------------------------------------------------------------------------
__global__ void embed_build_kernel(const float* __restrict__ emb, const int* __restrict__ xt,
                                   float* __restrict__ f, float* __restrict__ bb, int n)
{
    int i = blockIdx.x * blockDim.x + threadIdx.x; if (i >= n) return;
    int c = i % 768; int t = (i / 768) % 513; int b = i / (768 * 513);
    float fv = 0.f, bv = 0.f;
    if (t > 0) {
        fv = emb[(size_t)xt[b * 512 + t - 1] * 768 + c];
        bv = emb[(size_t)xt[b * 512 + 512 - t] * 768 + c];
    }
    f[i] = fv; bb[i] = bv;
}

__global__ void add_pos_cond_kernel(float* __restrict__ x, const float* __restrict__ pos,
                                    const float* __restrict__ cond, int Tt, int n)
{
    int i = blockIdx.x * blockDim.x + threadIdx.x; if (i >= n) return;
    int c = i % 768; int t = (i / 768) % Tt; int b = i / (768 * Tt);
    float v = x[i];
    if (pos)  v += pos[t * 768 + c];
    if (cond) v += cond[b * 768 + c];
    x[i] = v;
}

__global__ void add_head_pos_bf16_kernel(unsigned short* __restrict__ x,
                                         const float* __restrict__ pos, int n)
{
    int i = blockIdx.x * blockDim.x + threadIdx.x; if (i >= n) return;
    int c = i % 768; int t = (i / 768);
    float v = bf2f(x[i]) + pos[(t & 511) * 64 + (c & 63)];
    x[i] = f2bf(v);
}

__global__ void add3_kernel(float* __restrict__ d, const float* __restrict__ a,
                            const float* __restrict__ b, int n)
{
    int i = blockIdx.x * blockDim.x + threadIdx.x; if (i >= n) return;
    d[i] = a[i] + b[i];
}

__global__ void readout_prep_kernel(const float* __restrict__ f, const float* __restrict__ bb,
                                    const float* __restrict__ cond,
                                    float* __restrict__ cx, float* __restrict__ ae, int n)
{
    int i = blockIdx.x * blockDim.x + threadIdx.x; if (i >= n) return;
    int c = i % 768; int d = (i / 768) % 512; int b = i / (768 * 512);
    float fx = f[((size_t)b * 513 + d) * 768 + c];
    float bx = bb[((size_t)b * 513 + (511 - d)) * 768 + c];
    float cv = cond[b * 768 + c];
    cx[((size_t)b * 512 + d) * 768 + c] = (fx + bx) * 0.70710678118654752f + cv;
    ae[((size_t)b * 1024 + d) * 768 + c]       = fx + cv;
    ae[((size_t)b * 1024 + 512 + d) * 768 + c] = bx + cv;
}

__global__ void time_embed_kernel(const float* __restrict__ t, float* __restrict__ out)
{
    int b = blockIdx.x, j = threadIdx.x;
    float fr = __expf(-9.210340371976184f * (float)(j & 127) * (1.f / 128.f));
    float a = t[b] * fr;
    out[b * 256 + j] = (j < 128) ? cosf(a) : sinf(a);
}

__global__ __launch_bounds__(256) void final_sub_kernel(
    const float* __restrict__ h, const float* __restrict__ emb,
    const int* __restrict__ xt, float* __restrict__ logits)
{
    __shared__ float red[256];
    int row = blockIdx.x, tid = threadIdx.x;
    int b = row >> 9, d = row & 511;
    int tok = xt[b * 512 + d];
    const float* hr = h + (size_t)row * 768;
    const float* er = emb + (size_t)tok * 768;
    float s = 0.f;
    for (int i = tid; i < 768; i += 256) s += hr[i] * er[i];
    red[tid] = s; __syncthreads();
    for (int st = 128; st > 0; st >>= 1) { if (tid < st) red[tid] += red[tid + st]; __syncthreads(); }
    logits[(size_t)row * 256 + tid] -= red[0];
}

// ---------------------------------------------------------------------------
// Host side
// ---------------------------------------------------------------------------
struct Lin { const float* b; const float* w; const unsigned short* wt; };
struct LNP { const float* b; const float* g; };
struct MLPp { Lin fc1; Lin fc2; LNP ln; };
struct AttnP { LNP ln; MLPp mlp; Lin proj; Lin qkv; };
struct CausalP { AttnP layers[2]; const float* pos; Lin proj_in; };

extern "C" void kernel_launch(void* const* d_in, const int* in_sizes, int n_in,
                              void* d_out, int out_size, void* d_ws, size_t ws_size,
                              hipStream_t stream)
{
    (void)in_sizes; (void)n_in; (void)out_size; (void)ws_size;

    int idx = 0;
    auto F      = [&]() { return (const float*)d_in[idx++]; };
    auto getLin = [&]() { Lin l; l.b = F(); l.w = F(); l.wt = nullptr; return l; };
    auto getLN  = [&]() { LNP l; l.b = F(); l.g = F(); return l; };
    auto getMLP = [&]() { MLPp m; m.fc1 = getLin(); m.fc2 = getLin(); m.ln = getLN(); return m; };
    auto getAttn = [&]() { AttnP a; a.ln = getLN(); a.mlp = getMLP(); a.proj = getLin(); a.qkv = getLin(); return a; };
    auto getCausal = [&]() {
        CausalP c; c.layers[0] = getAttn(); c.layers[1] = getAttn();
        c.pos = F(); c.proj_in = getLin(); return c;
    };

    // alphabetical pytree leaf order: cond_t, params{bwd,fwd,out,readout,time,vocab_emb}, xt
    const float* cond_t = F();
    CausalP bwdP[2]; bwdP[0] = getCausal(); bwdP[1] = getCausal();
    CausalP fwdP[2]; fwdP[0] = getCausal(); fwdP[1] = getCausal();
    const float* out_emb = F(); LNP out_ln = getLN();
    Lin r_k = getLin(); LNP r_ln1 = getLN(); LNP r_ln2 = getLN();
    MLPp r_mlp = getMLP(); Lin r_out = getLin(); const float* r_pos = F();
    Lin r_q = getLin(); Lin r_v = getLin();
    Lin t_fc1 = getLin(); Lin t_fc2 = getLin();
    const float* vocab_emb = F();
    const int* xt = (const int*)d_in[idx++];

    const int Bb = 8, T = 513, Dd = 512, H = 768;
    const int MT = Bb * T;        // 4104
    const int MD = Bb * Dd;       // 4096
    const size_t nBT = (size_t)MT * H;
    const size_t nBD = (size_t)MD * H;

    // ---- float workspace arena ----
    size_t off = 0;
    auto alloc = [&](size_t n) { float* p = (float*)d_ws + off; off += n; return p; };
    float* cond = alloc(8 * 768);
    float* temb = alloc(8 * 256);
    float* tmid = alloc(8 * 768);
    float* W0  = alloc(nBT);                     // fwd activations
    float* W1  = alloc(nBT);                     // bwd activations
    float* W2  = alloc(nBT);                     // ping
    float* W3  = alloc(nBT);                     // LN / attn-out scratch
    float* W4  = alloc((size_t)MT * 2304);       // qkv bf16 / readout v bf16 (reinterpreted)
    float* W5  = alloc((size_t)MT * 3072);       // mlp hidden / aeln alias
    float* W6  = alloc(nBT);                     // residual save
    float* W7  = alloc(nBD);                     // cx
    float* W8  = alloc(nBD);                     // readout q (bf16 reinterpret)
    float* W9  = alloc(nBT);                     // pong
    float* W10 = alloc((size_t)Bb * 2 * Dd * H); // ae -> readout k (bf16 reinterpret)
    float* W11 = W5;                             // aeln aliases mlp-hidden region

    // ---- bf16 weight arena (converted once per launch; deterministic) ----
    unsigned short* ubase = (unsigned short*)((float*)d_ws + off);
    size_t uoff = 0;
    auto allocU = [&](size_t n) { unsigned short* p = ubase + uoff; uoff += n; return p; };
    auto wconv = [&](Lin& l, int K, int N) {
        unsigned short* dst = allocU((size_t)K * N);
        int n = K * N;
        wconv_kernel<<<(n + 255) / 256, 256, 0, stream>>>(l.w, dst, K, N);
        l.wt = dst;
    };
    wconv(t_fc1, 256, 768); wconv(t_fc2, 768, 768);
    for (CausalP* cp : { &fwdP[0], &fwdP[1], &bwdP[0], &bwdP[1] }) {
        wconv(cp->proj_in, 768, 768);
        for (int l = 0; l < 2; l++) {
            wconv(cp->layers[l].qkv, 768, 2304);
            wconv(cp->layers[l].proj, 768, 768);
            wconv(cp->layers[l].mlp.fc1, 768, 3072);
            wconv(cp->layers[l].mlp.fc2, 3072, 768);
        }
    }
    wconv(r_q, 768, 768); wconv(r_k, 768, 768); wconv(r_v, 768, 768); wconv(r_out, 768, 768);
    wconv(r_mlp.fc1, 768, 3072); wconv(r_mlp.fc2, 3072, 768);
    unsigned short* embU = allocU((size_t)256 * 768);   // already [N,K]
    conv_kernel<<<(256 * 768 + 255) / 256, 256, 0, stream>>>(out_emb, embU, 256 * 768);

    auto gemm = [&](const float* A, const Lin& l, const float* res,
                    void* C, int M, int N, int K, int act, int obf) {
        dim3 g(N / 64, (M + 63) / 64);
        gemm_kernel<<<g, 256, 0, stream>>>(A, l.wt, l.b, res, C, M, N, K, act, obf);
    };
    auto ln = [&](const float* x, LNP p, float* y, int rows) {
        layernorm_kernel<<<rows, 256, 0, stream>>>(x, p.g, p.b, y, 768);
    };
    auto ew = [&](int n) { return (n + 255) / 256; };

    // ---- timestep embedding -> cond [8,768] ----
    time_embed_kernel<<<8, 256, 0, stream>>>(cond_t, temb);
    gemm(temb, t_fc1, nullptr, tmid, 8, 768, 256, 2, 0);  // SiLU
    gemm(tmid, t_fc2, nullptr, cond, 8, 768, 768, 0, 0);

    // ---- token embeddings: f = [pad,x], b = flip([x,pad]) ----
    embed_build_kernel<<<ew((int)nBT), 256, 0, stream>>>(vocab_emb, xt, W0, W1, (int)nBT);

    // ---- causal transformer blocks ----
    auto run_causal = [&](float* x, const CausalP& cp) {
        hipMemcpyAsync(W6, x, nBT * sizeof(float), hipMemcpyDeviceToDevice, stream);
        gemm(x, cp.proj_in, nullptr, W2, MT, 768, 768, 0, 0);
        add_pos_cond_kernel<<<ew((int)nBT), 256, 0, stream>>>(W2, cp.pos, cond, T, (int)nBT);
        float* cur = W2; float* alt = W9;
        unsigned short* qkvU = (unsigned short*)W4;
        for (int l = 0; l < 2; l++) {
            const AttnP& lp = cp.layers[l];
            ln(cur, lp.ln, W3, MT);
            gemm(W3, lp.qkv, nullptr, qkvU, MT, 2304, 768, 0, 1);     // bf16 out
            attn_kernel<<<dim3(33, 12, 8), 32, 0, stream>>>(
                qkvU, qkvU + 768, qkvU + 1536, W3, T, T,
                2304, (long long)T * 2304, 2304, (long long)T * 2304,
                2304, (long long)T * 2304, 768, (long long)T * 768,
                0.125f, 0, Dd);
            gemm(W3, lp.proj, cur, alt, MT, 768, 768, 0, 0);
            { float* t = cur; cur = alt; alt = t; }
            ln(cur, lp.mlp.ln, W3, MT);
            gemm(W3, lp.mlp.fc1, nullptr, W5, MT, 3072, 768, 1, 0);   // GELU
            gemm(W5, lp.mlp.fc2, cur, alt, MT, 768, 3072, 0, 0);
            { float* t = cur; cur = alt; alt = t; }
        }
        add3_kernel<<<ew((int)nBT), 256, 0, stream>>>(x, cur, W6, (int)nBT);
    };

    run_causal(W0, fwdP[0]); run_causal(W0, fwdP[1]);
    run_causal(W1, bwdP[0]); run_causal(W1, bwdP[1]);

    // ---- readout ----
    readout_prep_kernel<<<ew((int)nBD), 256, 0, stream>>>(W0, W1, cond, W7, W10, (int)nBD);
    ln(W7, r_ln1, W2, MD);                  // cx -> ln
    ln(W10, r_ln1, W11, 2 * MD);            // ae -> ln (aliased into W5 region)
    unsigned short* qU = (unsigned short*)W8;
    unsigned short* kU = (unsigned short*)W10;
    unsigned short* vU = (unsigned short*)W4;
    gemm(W2,  r_q, nullptr, qU, MD,     768, 768, 0, 1);
    gemm(W11, r_k, nullptr, kU, 2 * MD, 768, 768, 0, 1);
    gemm(W11, r_v, nullptr, vU, 2 * MD, 768, 768, 0, 1);
    add_head_pos_bf16_kernel<<<ew((int)nBD), 256, 0, stream>>>(qU, r_pos, (int)nBD);
    add_head_pos_bf16_kernel<<<ew((int)(2 * nBD)), 256, 0, stream>>>(kU, r_pos, (int)(2 * nBD));
    attn_kernel<<<dim3(32, 12, 8), 32, 0, stream>>>(
        qU, kU, vU, W3, Dd, 2 * Dd,
        768, (long long)Dd * 768, 768, (long long)2 * Dd * 768,
        768, (long long)2 * Dd * 768, 768, (long long)Dd * 768,
        0.125f, 1, Dd);
    gemm(W3, r_out, W7, W9, MD, 768, 768, 0, 0);     // x = cx + aggregate
    ln(W9, r_ln2, W2, MD);
    ln(W2, r_mlp.ln, W3, MD);
    gemm(W3, r_mlp.fc1, nullptr, W5, MD, 3072, 768, 1, 0);  // GELU
    gemm(W5, r_mlp.fc2, W9, W6, MD, 768, 3072, 0, 0);       // h

    // ---- output head ----
    ln(W6, out_ln, W2, MD);
    add_pos_cond_kernel<<<ew((int)nBD), 256, 0, stream>>>(W2, nullptr, cond, Dd, (int)nBD);
    Lin embL; embL.b = nullptr; embL.w = out_emb; embL.wt = embU;
    gemm(W2, embL, nullptr, (float*)d_out, MD, 256, 768, 0, 0);   // h @ emb^T
    final_sub_kernel<<<MD, 256, 0, stream>>>(W2, out_emb, xt, (float*)d_out);
}
